// Zformer_23708219473921
// MI455X (gfx1250) — matrix operand order repaired and checked by reading, not compiled, not run on hardware
//
#include <hip/hip_runtime.h>

// ---------------------------------------------------------------------------
// Zformer linear-attention block for MI455X (gfx1250, wave32, WMMA).
// L=4096, B=4, D=1024, H=16, HD=64.
// ---------------------------------------------------------------------------

typedef __attribute__((ext_vector_type(16))) _Float16 v16h;
typedef __attribute__((ext_vector_type(8)))  _Float16 v8h;
typedef __attribute__((ext_vector_type(4)))  _Float16 v4h;
typedef __attribute__((ext_vector_type(8)))  float    v8f;

#define L_SEQ    4096
#define BATCH    4
#define DMODEL   1024
#define NHEAD    16
#define HDIM     64
#define Q_SCALE  0.125f          /* hd^-0.5 */
#define INV_SCL  (1.0f / 700.0f)

// ----- WMMA fragment helpers (layouts per CDNA5 ISA 7.12.2) -----------------

// A-matrix 16x32 f16: lane l -> row M = l&15; g = l>>4 selects K-half.
// elements 0..7  -> K = kk + 8*g + j
// elements 8..15 -> K = kk + 16 + 8*g + (j-8)
static __device__ __forceinline__ v16h load_a_frag(const _Float16* row, int kk, int g) {
    v8h lo = *(const v8h*)(row + kk + 8 * g);
    v8h hi = *(const v8h*)(row + kk + 16 + 8 * g);
    return __builtin_shufflevector(lo, hi, 0,1,2,3,4,5,6,7,8,9,10,11,12,13,14,15);
}

// B-matrix 32x16 f16: lane l -> col N = l&15; g = l>>4 selects K-half.
// elements 0..7 -> K = kk + 8*g + j; elements 8..15 -> K = kk + 16 + 8*g + (j-8).
static __device__ __forceinline__ v16h load_b_frag(const _Float16* row, int kk, int g) {
    v8h lo = *(const v8h*)(row + kk + 8 * g);
    v8h hi = *(const v8h*)(row + kk + 16 + 8 * g);
    return __builtin_shufflevector(lo, hi, 0,1,2,3,4,5,6,7,8,9,10,11,12,13,14,15);
}

static __device__ __forceinline__ v8f wmma_f16(v16h a, v16h b, v8f c) {
    return __builtin_amdgcn_wmma_f32_16x16x32_f16(false, a, false, b, (short)0, c,
                                                  false, false);
}

// ----- utility kernels ------------------------------------------------------

__global__ __launch_bounds__(256) void k_zero_f32(float* p, int n) {
    int i = blockIdx.x * 256 + threadIdx.x;
    if (i < n) p[i] = 0.0f;
}

__global__ __launch_bounds__(256) void k_cvt_w16(const float* __restrict__ src,
                                                 _Float16* __restrict__ dst, int n) {
    int i = blockIdx.x * 256 + threadIdx.x;
    if (i < n) dst[i] = (_Float16)src[i];
}

// kv32[bh][d][e] (fp32) -> kvT16[bh][e][d] (f16)  (B-fragment friendly)
__global__ __launch_bounds__(256) void k_cvt_kv(const float* __restrict__ kv32,
                                                _Float16* __restrict__ kvT16) {
    int i  = blockIdx.x * 256 + threadIdx.x;      // n = 64 * 4096
    int bh = i >> 12;
    int r  = i & 4095;
    int e  = r >> 6;
    int d  = r & 63;
    kvT16[i] = (_Float16)kv32[(bh << 12) + (d << 6) + e];
}

// ----- kernel 2: K/V projection + kv accumulation ---------------------------
// One WG = 128 tokens of one batch.  LDS: Xs 256KB + KsT 16KB + VsT 16KB.
__global__ __launch_bounds__(256) void k_kv_pass(const float* __restrict__ X,
        const _Float16* __restrict__ Wk16, const float* __restrict__ bk,
        const _Float16* __restrict__ Wv16, const float* __restrict__ bv,
        float* __restrict__ kv32) {
    extern __shared__ char smem[];
    _Float16* Xs  = (_Float16*)smem;                       // [128][1024]
    _Float16* KsT = (_Float16*)(smem + 128 * 1024 * 2);    // [64 feat][128 tok]
    _Float16* VsT = KsT + 64 * 128;                        // [64 feat][128 tok]

    const int tid  = threadIdx.x;
    const int wave = tid >> 5;
    const int lane = tid & 31;
    const int g    = lane >> 4;
    const int nn   = lane & 15;
    const int bb   = blockIdx.y;
    const int l0   = blockIdx.x * 128;

    // stage X tile (fp32 -> f16)
    for (int i = tid; i < 128 * 1024 / 4; i += 256) {
        int r  = i >> 8;
        int c4 = (i & 255) * 4;
        const float4 f = *(const float4*)(X + ((size_t)(l0 + r) * BATCH + bb) * DMODEL + c4);
        v4h hv = { (_Float16)f.x, (_Float16)f.y, (_Float16)f.z, (_Float16)f.w };
        *(v4h*)(Xs + r * 1024 + c4) = hv;
    }
    __syncthreads();

    const bool isV          = (wave >= 4);
    const int  nb           = wave & 3;                    // 16-col strip within head
    const _Float16* Wm      = isV ? Wv16 : Wk16;
    const float*    bias    = isV ? bv : bk;
    _Float16*       stT     = isV ? VsT : KsT;
    const int       fLoc    = nb * 16 + nn;

    for (int h = 0; h < NHEAD; ++h) {
        // ---- phase A: K_h / V_h projection (128 x 64 each) ----
        const _Float16* wrow = Wm + (size_t)(h * HDIM + fLoc) * DMODEL;
        const float     bval = bias[h * HDIM + fLoc];
        v8f acc[8] = {};
        for (int kk = 0; kk < DMODEL; kk += 32) {
            const v16h bf = load_b_frag(wrow, kk, g);
#pragma unroll
            for (int mb = 0; mb < 8; ++mb) {
                const v16h af = load_a_frag(Xs + (16 * mb + nn) * 1024, kk, g);
                acc[mb] = wmma_f16(af, bf, acc[mb]);
            }
        }
        // store transposed [feat][token]; C lane n -> feat, VGPR i -> token row,
        // so 8 consecutive i map to contiguous tokens: one 16B store.
#pragma unroll
        for (int mb = 0; mb < 8; ++mb) {
            v8h hv;
#pragma unroll
            for (int i = 0; i < 8; ++i) hv[i] = (_Float16)(acc[mb][i] + bval);
            *(v8h*)(stT + fLoc * 128 + 16 * mb + 8 * g) = hv;
        }
        __syncthreads();

        // ---- phase B: kv_h += K_hᵀ V_h  (64x64, K-dim = 128 tokens) ----
        {
            const int db  = wave & 3;
            const int eb0 = 2 * (wave >> 2);
            v8f acc2[2] = {};
            for (int kk = 0; kk < 128; kk += 32) {
                const v16h af = load_a_frag(KsT + (16 * db + nn) * 128, kk, g);
#pragma unroll
                for (int j = 0; j < 2; ++j) {
                    const v16h bf = load_b_frag(VsT + (16 * (eb0 + j) + nn) * 128, kk, g);
                    acc2[j] = wmma_f16(af, bf, acc2[j]);
                }
            }
            float* kvh = kv32 + (size_t)(bb * NHEAD + h) * HDIM * HDIM;
#pragma unroll
            for (int j = 0; j < 2; ++j)
#pragma unroll
                for (int i = 0; i < 8; ++i) {
                    int d = 16 * db + i + 8 * g;
                    int e = 16 * (eb0 + j) + nn;
                    __hip_atomic_fetch_add(&kvh[d * HDIM + e], acc2[j][i],
                                           __ATOMIC_RELAXED, __HIP_MEMORY_SCOPE_AGENT);
                }
        }
        __syncthreads();   // protect stage buffers before next head
    }
}

// ----- kernel 3: fused Q -> attn -> output projection -----------------------
// One WG = 64 tokens of one batch.  LDS: buf0 (Xs, later attn feats) + buf1 (Q).
__global__ __launch_bounds__(256) void k_qo_pass(const float* __restrict__ X,
        const _Float16* __restrict__ Wq16, const float* __restrict__ bq,
        const _Float16* __restrict__ kvT16,
        const _Float16* __restrict__ Wo16, const float* __restrict__ bo,
        float* __restrict__ out) {
    extern __shared__ char smem[];
    _Float16* buf0 = (_Float16*)smem;          // [64][1024]: X, then attn features
    _Float16* buf1 = buf0 + 64 * 1024;         // [64][1024]: Q (f16, pre-scaled)

    const int tid  = threadIdx.x;
    const int wave = tid >> 5;
    const int lane = tid & 31;
    const int g    = lane >> 4;
    const int nn   = lane & 15;
    const int bb   = blockIdx.y;
    const int l0   = blockIdx.x * 64;

    for (int i = tid; i < 64 * 1024 / 4; i += 256) {
        int r  = i >> 8;
        int c4 = (i & 255) * 4;
        const float4 f = *(const float4*)(X + ((size_t)(l0 + r) * BATCH + bb) * DMODEL + c4);
        v4h hv = { (_Float16)f.x, (_Float16)f.y, (_Float16)f.z, (_Float16)f.w };
        *(v4h*)(buf0 + r * 1024 + c4) = hv;
    }
    __syncthreads();

    // ---- phase 2: Q = (X @ Wqᵀ + bq) * 0.125, all 1024 features ----
    for (int c = 0; c < 2; ++c) {
        const int nb0 = 8 * wave + 4 * c;      // wave owns cols [128w, 128w+128)
        v8f acc[4][4] = {};
        for (int kk = 0; kk < DMODEL; kk += 32) {
            v16h a[4];
#pragma unroll
            for (int mb = 0; mb < 4; ++mb)
                a[mb] = load_a_frag(buf0 + (16 * mb + nn) * 1024, kk, g);
#pragma unroll
            for (int j = 0; j < 4; ++j) {
                const v16h bf = load_b_frag(Wq16 + (size_t)(16 * (nb0 + j) + nn) * DMODEL, kk, g);
#pragma unroll
                for (int mb = 0; mb < 4; ++mb)
                    acc[mb][j] = wmma_f16(a[mb], bf, acc[mb][j]);
            }
        }
#pragma unroll
        for (int j = 0; j < 4; ++j) {
            const int   f  = 16 * (nb0 + j) + nn;
            const float bv = bq[f];
#pragma unroll
            for (int mb = 0; mb < 4; ++mb)
#pragma unroll
                for (int i = 0; i < 8; ++i) {
                    int m = 16 * mb + i + 8 * g;
                    buf1[m * 1024 + f] = (_Float16)((acc[mb][j][i] + bv) * Q_SCALE);
                }
        }
    }
    __syncthreads();

    // ---- phase 3: attn_h = Q_h @ kv_h / 700, written in (hd, h) interleave ----
    {
        const int mb  = wave & 3;
        const int eb0 = 2 * (wave >> 2);
        for (int h = 0; h < NHEAD; ++h) {
            v8f acc2[2] = {};
            for (int kk = 0; kk < HDIM; kk += 32) {
                const v16h a = load_a_frag(buf1 + (16 * mb + nn) * 1024 + h * HDIM, kk, g);
#pragma unroll
                for (int j = 0; j < 2; ++j) {
                    const v16h bf = load_b_frag(
                        kvT16 + ((size_t)(bb * NHEAD + h) * HDIM + 16 * (eb0 + j) + nn) * HDIM,
                        kk, g);
                    acc2[j] = wmma_f16(a, bf, acc2[j]);
                }
            }
#pragma unroll
            for (int j = 0; j < 2; ++j)
#pragma unroll
                for (int i = 0; i < 8; ++i) {
                    int m = 16 * mb + i + 8 * g;
                    int e = 16 * (eb0 + j) + nn;
                    buf0[m * 1024 + e * NHEAD + h] = (_Float16)(acc2[j][i] * INV_SCL);
                }
        }
    }
    __syncthreads();

    // ---- phase 4: out = attn_feats @ Woᵀ + bo, store (l, b, d) fp32 ----
    for (int c = 0; c < 2; ++c) {
        const int nb0 = 8 * wave + 4 * c;
        v8f acc[4][4] = {};
        for (int kk = 0; kk < DMODEL; kk += 32) {
            v16h a[4];
#pragma unroll
            for (int mb = 0; mb < 4; ++mb)
                a[mb] = load_a_frag(buf0 + (16 * mb + nn) * 1024, kk, g);
#pragma unroll
            for (int j = 0; j < 4; ++j) {
                const v16h bf = load_b_frag(Wo16 + (size_t)(16 * (nb0 + j) + nn) * DMODEL, kk, g);
#pragma unroll
                for (int mb = 0; mb < 4; ++mb)
                    acc[mb][j] = wmma_f16(a[mb], bf, acc[mb][j]);
            }
        }
#pragma unroll
        for (int j = 0; j < 4; ++j) {
            const int   f  = 16 * (nb0 + j) + nn;
            const float bv = bo[f];
#pragma unroll
            for (int mb = 0; mb < 4; ++mb)
#pragma unroll
                for (int i = 0; i < 8; ++i) {
                    int m = 16 * mb + i + 8 * g;
                    out[((size_t)(l0 + m) * BATCH + bb) * DMODEL + f] = acc[mb][j][i] + bv;
                }
        }
    }
}

// ---------------------------------------------------------------------------

extern "C" void kernel_launch(void* const* d_in, const int* in_sizes, int n_in,
                              void* d_out, int out_size, void* d_ws, size_t ws_size,
                              hipStream_t stream) {
    const float* X  = (const float*)d_in[0];
    const float* Wq = (const float*)d_in[2];
    const float* bq = (const float*)d_in[3];
    const float* Wk = (const float*)d_in[4];
    const float* bk = (const float*)d_in[5];
    const float* Wv = (const float*)d_in[6];
    const float* bv = (const float*)d_in[7];
    const float* Wo = (const float*)d_in[8];
    const float* bo = (const float*)d_in[9];
    float* out = (float*)d_out;

    char* ws = (char*)d_ws;
    float*     kv32  = (float*)ws;                         // 1 MB
    _Float16*  kvT16 = (_Float16*)(ws + (2u << 20));       // 512 KB
    _Float16*  Wq16  = (_Float16*)(ws + (4u  << 20));      // 2 MB each
    _Float16*  Wk16  = (_Float16*)(ws + (6u  << 20));
    _Float16*  Wv16  = (_Float16*)(ws + (8u  << 20));
    _Float16*  Wo16  = (_Float16*)(ws + (10u << 20));

    const int kvN = BATCH * NHEAD * HDIM * HDIM;           // 262144
    const int wN  = DMODEL * DMODEL;                       // 1048576

    k_zero_f32<<<(kvN + 255) / 256, 256, 0, stream>>>(kv32, kvN);
    k_cvt_w16<<<(wN + 255) / 256, 256, 0, stream>>>(Wq, Wq16, wN);
    k_cvt_w16<<<(wN + 255) / 256, 256, 0, stream>>>(Wk, Wk16, wN);
    k_cvt_w16<<<(wN + 255) / 256, 256, 0, stream>>>(Wv, Wv16, wN);
    k_cvt_w16<<<(wN + 255) / 256, 256, 0, stream>>>(Wo, Wo16, wN);

    // KV pass: 128-token tiles, LDS = 256KB (X) + 32KB (K/V stages)
    const size_t lds_kv = (size_t)128 * 1024 * 2 + 2 * 64 * 128 * 2;
    k_kv_pass<<<dim3(L_SEQ / 128, BATCH), 256, lds_kv, stream>>>(X, Wk16, bk, Wv16, bv, kv32);

    k_cvt_kv<<<(kvN + 255) / 256, 256, 0, stream>>>(kv32, kvT16);

    // Q/attn/output pass: 64-token tiles, LDS = 2 x 128KB
    const size_t lds_qo = (size_t)2 * 64 * 1024 * 2;
    k_qo_pass<<<dim3(L_SEQ / 64, BATCH), 256, lds_qo, stream>>>(X, Wq16, bq, kvT16,
                                                                Wo16, bo, out);
}
